// DynamicKernelUpsamplerAniso_28922309771528
// MI455X (gfx1250) — matrix operand order repaired and checked by reading, not compiled, not run on hardware
//
#include <hip/hip_runtime.h>

typedef __attribute__((ext_vector_type(16))) _Float16 v16h;
typedef __attribute__((ext_vector_type(8)))  float    v8f;

#define TILE     128          // pixels per workgroup (one row segment)
#define NTHREADS 256          // 8 wave32 -> 8 M-tiles of 16 pixels
#define HID      64           // hidden channels
#define NCH      80           // 75 kernel-predictor channels padded to 80
#define KK       25           // 5x5 reassembly taps
#define HALO_W   (TILE + 4)   // +-2 halo for the 5x5 unfold

__launch_bounds__(NTHREADS, 1)
__global__ void carafe_fused_wmma(const float* __restrict__ x,
                                  const float* __restrict__ w1,
                                  const float* __restrict__ b1,
                                  const float* __restrict__ w2,
                                  const float* __restrict__ b2,
                                  const float* __restrict__ wp,
                                  const float* __restrict__ bp,
                                  float* __restrict__ out,
                                  int B, int H, int W) {
  __shared__ float    s_xg[5 * HALO_W];    // gray tile + halo (rows h-2..h+2)
  __shared__ _Float16 s_w1h[HID * 16];     // w1 as [n][tap], taps padded 9->16
  __shared__ _Float16 s_w2h[NCH * HID];    // w2 as [c][k], c padded 75->80
  __shared__ float    s_b1[HID];
  __shared__ float    s_b2[NCH];
  __shared__ _Float16 s_h[TILE * HID];     // hidden activations (post-ReLU)
  __shared__ float    s_ker[TILE * NCH];   // kernel-predictor logits

  const int tid = threadIdx.x;
  const int w0  = blockIdx.x * TILE;
  const int h   = blockIdx.y;
  const int b   = blockIdx.z;

  // ---------------- Phase 0: stage weights into LDS as f16 ----------------
  for (int i = tid; i < HID * 16; i += NTHREADS) {
    int t = i & 15;
    s_w1h[i] = (_Float16)((t < 9) ? w1[(i >> 4) * 9 + t] : 0.0f);
  }
  for (int i = tid; i < NCH * HID; i += NTHREADS) {
    int c = i >> 6;
    s_w2h[i] = (_Float16)((c < 75) ? w2[i] : 0.0f);
  }
  for (int i = tid; i < NCH; i += NTHREADS) s_b2[i] = (i < 75) ? b2[i] : 0.0f;
  for (int i = tid; i < HID; i += NTHREADS) s_b1[i] = b1[i];

  // ---------------- Phase 1: gray tile (channel mean) with halo -----------
  const size_t plane = (size_t)H * W;
  const float* xb = x + (size_t)b * 3 * plane;
  for (int i = tid; i < 5 * HALO_W; i += NTHREADS) {
    int r  = i / HALO_W;
    int c  = i - r * HALO_W;
    int gr = h + r - 2;
    int gc = w0 + c - 2;
    float v = 0.0f;
    if (gr >= 0 && gr < H && gc >= 0 && gc < W) {
      size_t o = (size_t)gr * W + gc;
      v = (xb[o] + xb[plane + o] + xb[2 * plane + o]) * (1.0f / 3.0f);
    }
    s_xg[i] = v;
  }
  __syncthreads();

  const int wave = tid >> 5;
  const int lane = tid & 31;
  const int hs   = lane >> 4;          // which K-half this lane carries (A/B layout)
  const int ln   = lane & 15;
  const int prow = wave * 16 + ln;     // pixel row (M) for A operands

  // ---------------- Phase 2: conv3x3 + ReLU via WMMA ----------------------
  // A: 16x32 f16, K = 9 taps zero-padded to 32.
  // Layout: lanes 0-15 -> K0..15; lanes 16-31 -> K16..31 (contiguous, matches B)
  v16h a;
  #pragma unroll
  for (int j = 0; j < 16; ++j) a[j] = (_Float16)0.0f;
  if (hs == 0) {
    #pragma unroll
    for (int j = 0; j < 8; ++j) {            // taps 0..7
      int ky = j / 3, kx = j - ky * 3;
      a[j] = (_Float16)s_xg[(ky + 1) * HALO_W + (prow + kx + 1)];
    }
    a[8] = (_Float16)s_xg[3 * HALO_W + (prow + 3)];   // tap 8 = K8 (ky=2,kx=2)
  }

  #pragma unroll
  for (int t = 0; t < 4; ++t) {              // 64 hidden channels = 4 N-tiles
    int n = t * 16 + ln;
    v16h bm;
    #pragma unroll
    for (int j = 0; j < 16; ++j) bm[j] = (_Float16)0.0f;
    if (hs == 0) {                            // B lanes 0-15 carry K0..15
      #pragma unroll
      for (int j = 0; j < 16; ++j) bm[j] = s_w1h[n * 16 + j];
    }                                         // lanes 16-31 carry K16..31 == 0
    float bias = s_b1[n];
    v8f acc;
    #pragma unroll
    for (int r = 0; r < 8; ++r) acc[r] = bias;
    acc = __builtin_amdgcn_wmma_f32_16x16x32_f16(false, a, false, bm,
                                                 (short)0, acc, false, false);
    #pragma unroll
    for (int r = 0; r < 8; ++r) {             // D: row = r + 8*hs, N = ln
      int p = wave * 16 + r + hs * 8;
      float z = acc[r];
      s_h[p * HID + n] = (_Float16)(z > 0.0f ? z : 0.0f);   // ReLU, no canonicalize
    }
  }
  __syncthreads();

  // ---------------- Phase 3: 1x1 conv (64 -> 80) via WMMA ------------------
  // A: hidden acts, two K-windows of 32 each.
  v16h a0, a1;
  {
    int base0 = prow * HID + hs * 16;         // window s=0
    #pragma unroll
    for (int j = 0; j < 8; ++j) {
      a0[j]     = s_h[base0 + j];
      a0[8 + j] = s_h[base0 + 8 + j];
    }
    int base1 = base0 + 32;                   // window s=1
    #pragma unroll
    for (int j = 0; j < 8; ++j) {
      a1[j]     = s_h[base1 + j];
      a1[8 + j] = s_h[base1 + 8 + j];
    }
  }

  #pragma unroll
  for (int t = 0; t < 5; ++t) {              // 80 channels = 5 N-tiles
    int ch = t * 16 + ln;
    int wb = ch * HID + hs * 16;             // B: lanes 0-15 K0..15, 16-31 K16..31
    v16h b0, b1v;
    #pragma unroll
    for (int j = 0; j < 16; ++j) b0[j]  = s_w2h[wb + j];
    #pragma unroll
    for (int j = 0; j < 16; ++j) b1v[j] = s_w2h[wb + 32 + j];
    float bias = s_b2[ch];
    v8f acc;
    #pragma unroll
    for (int r = 0; r < 8; ++r) acc[r] = bias;
    acc = __builtin_amdgcn_wmma_f32_16x16x32_f16(false, a0, false, b0,
                                                 (short)0, acc, false, false);
    acc = __builtin_amdgcn_wmma_f32_16x16x32_f16(false, a1, false, b1v,
                                                 (short)0, acc, false, false);
    #pragma unroll
    for (int r = 0; r < 8; ++r) {
      int p = wave * 16 + r + hs * 8;
      s_ker[p * NCH + ch] = acc[r];
    }
  }
  __syncthreads();

  // ---------------- Phase 4: softmax over 25 taps + reassembly -------------
  const float wscale = wp[0];
  const float wbias  = bp[0];
  const size_t outW     = (size_t)W * 3;       // 1536
  const size_t chstride = (size_t)H * outW;
  for (int item = tid; item < TILE * 3; item += NTHREADS) {
    int p = item / 3;
    int v = item - p * 3;
    // Stage the 25 strided logits into registers once (used by both passes).
    float l[KK];
    const float* Lp = &s_ker[p * NCH + v];
    #pragma unroll
    for (int k = 0; k < KK; ++k) l[k] = Lp[k * 3];
    float m = l[0];
    #pragma unroll
    for (int k = 1; k < KK; ++k) m = (l[k] > m) ? l[k] : m;
    float ssum = 0.0f, wsum = 0.0f;
    #pragma unroll
    for (int k = 0; k < KK; ++k) {
      float e = __expf(l[k] - m);
      int ky = k / 5, kx = k - ky * 5;
      ssum += e;
      wsum += e * s_xg[ky * HALO_W + (p + kx)];  // xg[h+ky-2][w0+p+kx-2]
    }
    float o = (wsum / ssum) * wscale + wbias;
    size_t base = ((size_t)b * 3) * chstride + (size_t)h * outW
                + (size_t)(w0 + p) * 3 + v;
    // Streaming (non-temporal) stores: output is never re-read; keep L2 for
    // the x halo rows and weights instead.
    __builtin_nontemporal_store(o, &out[base]);
    __builtin_nontemporal_store(o, &out[base + chstride]);
    __builtin_nontemporal_store(o, &out[base + 2 * chstride]);
  }
}

extern "C" void kernel_launch(void* const* d_in, const int* in_sizes, int n_in,
                              void* d_out, int out_size, void* d_ws, size_t ws_size,
                              hipStream_t stream) {
  const float* x  = (const float*)d_in[0];
  const float* w1 = (const float*)d_in[1];
  const float* b1 = (const float*)d_in[2];
  const float* w2 = (const float*)d_in[3];
  const float* b2 = (const float*)d_in[4];
  const float* wp = (const float*)d_in[5];
  const float* bp = (const float*)d_in[6];
  float* out = (float*)d_out;

  const int H = 512, W = 512;
  const int B = in_sizes[0] / (3 * H * W);   // 4

  dim3 grid(W / TILE, H, B);                 // 4 x 512 x 4 workgroups
  carafe_fused_wmma<<<grid, NTHREADS, 0, stream>>>(x, w1, b1, w2, b2, wp, bp,
                                                   out, B, H, W);
}
